// ReneMHA_18657337934158
// MI455X (gfx1250) — matrix operand order repaired and checked by reading, not compiled, not run on hardware
//
#include <hip/hip_runtime.h>

typedef __attribute__((ext_vector_type(16))) __bf16 v16bf;
typedef __attribute__((ext_vector_type(8)))  float  v8f;

#define S_LEN   2048
#define DMODEL  2048
#define NB      2
#define NH      32
#define NKV     8
#define HEADD   64
#define WINLEN  1024
#define QKVDIM  3072

// ---------------------------------------------------------------------------
// fp32 -> bf16 conversion (grid-stride)
// ---------------------------------------------------------------------------
__global__ void cvt_f32_bf16(const float* __restrict__ src,
                             __bf16* __restrict__ dst, int n) {
  int i = blockIdx.x * blockDim.x + threadIdx.x;
  int stride = gridDim.x * blockDim.x;
  for (; i < n; i += stride) dst[i] = (__bf16)src[i];
}

// ---------------------------------------------------------------------------
// Wave-level row reductions (wave32; xor masks < 16 stay within 16-lane half,
// matching the WMMA C-layout where a 16-lane half covers all N for fixed M)
// ---------------------------------------------------------------------------
__device__ __forceinline__ float rowmax16(float v) {
  #pragma unroll
  for (int m = 1; m < 16; m <<= 1) v = fmaxf(v, __shfl_xor(v, m, 32));
  return v;
}
__device__ __forceinline__ float rowsum16(float v) {
  #pragma unroll
  for (int m = 1; m < 16; m <<= 1) v += __shfl_xor(v, m, 32);
  return v;
}

// Async global -> LDS DMA (CDNA5, ASYNCcnt-tracked). Per-lane: 16 bytes from
// `gaddr` into LDS byte offset `ldsoff` (low 32 bits of generic LDS pointer,
// as the flat LDS aperture uses the low address bits).
__device__ __forceinline__ void async_ld_b128(unsigned ldsoff,
                                              unsigned long long gaddr) {
  asm volatile("global_load_async_to_lds_b128 %0, %1, off"
               :: "v"(ldsoff), "v"(gaddr) : "memory");
}
__device__ __forceinline__ void wait_async0() {
  asm volatile("s_wait_asynccnt 0x0" ::: "memory");
}

// ---------------------------------------------------------------------------
// Tiled bf16 GEMM:  C[m][n] = sum_k A[m][k] * Bm[n][k] + bias[n]
//   (both operands K-contiguous; Bm rows are output features)
// Block tile 128x128, K-step 32, 256 threads = 8 waves (4x2), each wave
// computes 32x64 via 2x4 v_wmma_f32_16x16x32_bf16 accumulators. Tiles are
// staged with async global->LDS DMA (double buffered), fenced by
// s_wait_asynccnt + barrier, so the kt+1 DMA overlaps the kt WMMA stream.
// EPI==0: write fp32 to Cout[M][N]
// EPI==1: fused QKV split: scatter bf16 into q[b,h,s,d], k[b,g,s,d], vT[b,g,d,s]
// ---------------------------------------------------------------------------
template <int EPI>
__global__ __launch_bounds__(256) void gemm_bf16(
    const __bf16* __restrict__ A, const __bf16* __restrict__ Bm,
    const float* __restrict__ bias, float* __restrict__ Cout,
    __bf16* __restrict__ qb, __bf16* __restrict__ kb, __bf16* __restrict__ vTb,
    int M, int N, int K) {
  // 40 bf16 row stride = 80 bytes: 16B-aligned chunks + conflict-free banks
  __shared__ __bf16 lA[2][128][40];
  __shared__ __bf16 lB[2][128][40];

  const int tid = threadIdx.x;
  const int n0 = blockIdx.x * 128;
  const int m0 = blockIdx.y * 128;
  const int wave = tid >> 5, lane = tid & 31;
  const int half = lane >> 4, lr = lane & 15;
  const int wm = wave >> 1, wn = wave & 1;

  v8f acc[2][4];
  #pragma unroll
  for (int tm = 0; tm < 2; ++tm)
    #pragma unroll
    for (int tn = 0; tn < 4; ++tn) acc[tm][tn] = v8f{};

  const int nk = K >> 5;

  auto loadTileAsync = [&](int kt, int buf) {
    const int k0 = kt << 5;
    #pragma unroll
    for (int it = 0; it < 2; ++it) {
      int idx = tid + (it << 8);
      int row = idx >> 2, q = idx & 3;  // 4 x 8-bf16 chunks per 32-wide row
      async_ld_b128(
          (unsigned)(uintptr_t)&lA[buf][row][q * 8],
          (unsigned long long)(uintptr_t)(A + (size_t)(m0 + row) * K + k0 + q * 8));
      async_ld_b128(
          (unsigned)(uintptr_t)&lB[buf][row][q * 8],
          (unsigned long long)(uintptr_t)(Bm + (size_t)(n0 + row) * K + k0 + q * 8));
    }
  };

  loadTileAsync(0, 0);
  wait_async0();
  __syncthreads();

  for (int kt = 0; kt < nk; ++kt) {
    const int buf = kt & 1;
    if (kt + 1 < nk) loadTileAsync(kt + 1, buf ^ 1);
    if (kt + 2 < nk) {  // L2 prefetch of the tile after next
      __builtin_prefetch(A + (size_t)(m0 + (tid >> 1)) * K + ((kt + 2) << 5), 0, 3);
      __builtin_prefetch(Bm + (size_t)(n0 + (tid >> 1)) * K + ((kt + 2) << 5), 0, 3);
    }

    // A fragments: lane = row; lanes 0-15 hold K 0..7 & 16..23, lanes 16-31
    // hold K 8..15 & 24..31  -> chunks at [half*8] and [half*8+16]
    v16bf afrag[2];
    #pragma unroll
    for (int tm = 0; tm < 2; ++tm) {
      int row = wm * 32 + tm * 16 + lr;
      uint4* fp = reinterpret_cast<uint4*>(&afrag[tm]);
      fp[0] = *reinterpret_cast<const uint4*>(&lA[buf][row][half * 8]);
      fp[1] = *reinterpret_cast<const uint4*>(&lA[buf][row][half * 8 + 16]);
    }
    // B fragments: lane = column; same K split as A (K 0..7 & 16..23 / 8..15 & 24..31)
    v16bf bfrag[4];
    #pragma unroll
    for (int tn = 0; tn < 4; ++tn) {
      int row = wn * 64 + tn * 16 + lr;
      uint4* fp = reinterpret_cast<uint4*>(&bfrag[tn]);
      fp[0] = *reinterpret_cast<const uint4*>(&lB[buf][row][half * 8]);
      fp[1] = *reinterpret_cast<const uint4*>(&lB[buf][row][half * 8 + 16]);
    }

    #pragma unroll
    for (int tm = 0; tm < 2; ++tm)
      #pragma unroll
      for (int tn = 0; tn < 4; ++tn)
        acc[tm][tn] = __builtin_amdgcn_wmma_f32_16x16x32_bf16(
            false, afrag[tm], false, bfrag[tn], (short)0, acc[tm][tn], false, false);

    // publish next buffer: DMA (overlapped with the WMMAs above) must land
    // before any wave reads it after the barrier
    wait_async0();
    __syncthreads();
  }

  // epilogue: C layout -> lane holds column lr, VGPR r -> row r + 8*half
  #pragma unroll
  for (int tm = 0; tm < 2; ++tm)
    #pragma unroll
    for (int tn = 0; tn < 4; ++tn)
      #pragma unroll
      for (int r = 0; r < 8; ++r) {
        int row = m0 + wm * 32 + tm * 16 + r + half * 8;
        int col = n0 + wn * 64 + tn * 16 + lr;
        float val = acc[tm][tn][r] + bias[col];
        if (EPI == 0) {
          Cout[(size_t)row * N + col] = val;
        } else {
          int b = row >> 11, s = row & (S_LEN - 1);
          if (col < NH * HEADD) {
            int hh = col >> 6, d = col & 63;
            qb[(((size_t)(b * NH + hh)) * S_LEN + s) * HEADD + d] = (__bf16)val;
          } else {
            int e2 = col - NH * HEADD;
            int c = e2 >> 9, gg = (e2 >> 6) & 7, d = e2 & 63;
            if (c == 0)
              kb[(((size_t)(b * NKV + gg)) * S_LEN + s) * HEADD + d] = (__bf16)val;
            else
              vTb[(((size_t)(b * NKV + gg)) * HEADD + d) * S_LEN + s] = (__bf16)val;
          }
        }
      }
}

// ---------------------------------------------------------------------------
// Sliding-window flash attention. One wave per (batch, head, 16-query tile).
// Keys processed in 32-wide chunks: 4 WMMAs for Q.K^T, online softmax with
// cross-lane reductions, P staged via LDS (C-layout -> A-layout), 4 WMMAs
// for P.V using the pre-transposed V (K-contiguous B operand).
// ---------------------------------------------------------------------------
__global__ __launch_bounds__(128) void attn_swa(
    const __bf16* __restrict__ qb, const __bf16* __restrict__ kb,
    const __bf16* __restrict__ vTb, __bf16* __restrict__ ctxb) {
  __shared__ __bf16 ps[4][16][40];  // per-wave P staging (16 rows x 32 keys)

  const int wv = threadIdx.x >> 5, lane = threadIdx.x & 31;
  const int half = lane >> 4, lr = lane & 15;
  const int wid = blockIdx.x * 4 + wv;         // 8192 wave-tasks
  const int b = wid >> 12;                     // / (NH * S/16)
  const int rem = wid & 4095;
  const int h = rem >> 7, qt = rem & 127;
  const int s0 = qt << 4, g = h >> 2;          // 4 q-heads per kv head

  const __bf16* qh = qb + (((size_t)(b * NH + h)) * S_LEN + s0) * HEADD;
  const __bf16* kh = kb + ((size_t)(b * NKV + g)) * S_LEN * HEADD;
  const __bf16* vh = vTb + ((size_t)(b * NKV + g)) * HEADD * S_LEN;

  // Q fragments (A-layout), HD=64 -> two K-steps of 32
  v16bf qf[2];
  #pragma unroll
  for (int t = 0; t < 2; ++t) {
    uint4* fp = reinterpret_cast<uint4*>(&qf[t]);
    const __bf16* qp = qh + (size_t)lr * HEADD + t * 32 + half * 8;
    fp[0] = *reinterpret_cast<const uint4*>(qp);
    fp[1] = *reinterpret_cast<const uint4*>(qp + 16);
  }

  v8f acc[4];
  #pragma unroll
  for (int dt = 0; dt < 4; ++dt) acc[dt] = v8f{};
  float mrow[8], lrow[8];
  #pragma unroll
  for (int r = 0; r < 8; ++r) { mrow[r] = -__builtin_inff(); lrow[r] = 0.f; }

  int kmin = s0 + 15 - (WINLEN - 1);
  if (kmin < 0) kmin = 0;
  const int kc0 = kmin & ~31;
  const int kcl = (s0 + 15) & ~31;
  const float scale = 0.125f;  // 1/sqrt(64)

  for (int kc = kc0; kc <= kcl; kc += 32) {
    // scores: 16 queries x 32 keys = two 16x16 accumulators
    v8f sc[2] = {v8f{}, v8f{}};
    #pragma unroll
    for (int k2 = 0; k2 < 2; ++k2) {       // key half
      #pragma unroll
      for (int hs = 0; hs < 2; ++hs) {     // HD step
        v16bf kf;
        uint4* fp = reinterpret_cast<uint4*>(&kf);
        const __bf16* kp =
            kh + (size_t)(kc + k2 * 16 + lr) * HEADD + hs * 32 + half * 8;
        fp[0] = *reinterpret_cast<const uint4*>(kp);
        fp[1] = *reinterpret_cast<const uint4*>(kp + 16);
        sc[k2] = __builtin_amdgcn_wmma_f32_16x16x32_bf16(
            false, qf[hs], false, kf, (short)0, sc[k2], false, false);
      }
    }

    // online softmax per row (each 16-lane half owns rows r+8*half)
    #pragma unroll
    for (int r = 0; r < 8; ++r) {
      const int i = s0 + r + half * 8;
      const int j0 = kc + lr, j1 = j0 + 16;
      float v0 = (j0 <= i && j0 + WINLEN > i) ? sc[0][r] * scale : -__builtin_inff();
      float v1 = (j1 <= i && j1 + WINLEN > i) ? sc[1][r] * scale : -__builtin_inff();
      float mx = rowmax16(fmaxf(v0, v1));
      float mnew = fmaxf(mrow[r], mx);
      float alpha = __expf(mrow[r] - mnew);  // first iter: exp(-inf)=0
      float p0 = __expf(v0 - mnew);
      float p1 = __expf(v1 - mnew);
      float rs = rowsum16(p0 + p1);
      lrow[r] = lrow[r] * alpha + rs;
      mrow[r] = mnew;
      #pragma unroll
      for (int dt = 0; dt < 4; ++dt) acc[dt][r] *= alpha;
      ps[wv][r + half * 8][lr] = (__bf16)p0;
      ps[wv][r + half * 8][16 + lr] = (__bf16)p1;
    }

    // P (16x32, A-layout from LDS) x V (32 keys x 64 hd, K-contig from vT)
    v16bf pf;
    uint4* fp = reinterpret_cast<uint4*>(&pf);
    fp[0] = *reinterpret_cast<const uint4*>(&ps[wv][lr][half * 8]);
    fp[1] = *reinterpret_cast<const uint4*>(&ps[wv][lr][half * 8 + 16]);
    #pragma unroll
    for (int dt = 0; dt < 4; ++dt) {
      v16bf vf;
      uint4* vp = reinterpret_cast<uint4*>(&vf);
      const __bf16* vpp = vh + (size_t)(dt * 16 + lr) * S_LEN + kc + half * 8;
      vp[0] = *reinterpret_cast<const uint4*>(vpp);
      vp[1] = *reinterpret_cast<const uint4*>(vpp + 16);
      acc[dt] = __builtin_amdgcn_wmma_f32_16x16x32_bf16(
          false, pf, false, vf, (short)0, acc[dt], false, false);
    }
  }

  // normalize and write ctx bf16 [B][S][H*HD]
  #pragma unroll
  for (int dt = 0; dt < 4; ++dt)
    #pragma unroll
    for (int r = 0; r < 8; ++r) {
      int row = s0 + r + half * 8;
      int col = h * HEADD + dt * 16 + lr;
      ctxb[((size_t)(b * S_LEN) + row) * DMODEL + col] =
          (__bf16)(acc[dt][r] / lrow[r]);
    }
}

// ---------------------------------------------------------------------------
extern "C" void kernel_launch(void* const* d_in, const int* in_sizes, int n_in,
                              void* d_out, int out_size, void* d_ws,
                              size_t ws_size, hipStream_t stream) {
  (void)in_sizes; (void)n_in; (void)out_size; (void)ws_size;
  const float* x    = (const float*)d_in[0];
  const float* Wqkv = (const float*)d_in[1];
  const float* bqkv = (const float*)d_in[2];
  const float* Wout = (const float*)d_in[3];
  const float* bout = (const float*)d_in[4];
  float* out = (float*)d_out;

  const int M = NB * S_LEN;  // 4096 tokens
  char* ws = (char*)d_ws;
  size_t off = 0;
  auto carve = [&](size_t bytes) {
    void* p = ws + off;
    off += (bytes + 255) & ~(size_t)255;
    return p;
  };
  __bf16* xb    = (__bf16*)carve((size_t)M * DMODEL * 2);          // 16.8 MB
  __bf16* wqkvb = (__bf16*)carve((size_t)QKVDIM * DMODEL * 2);     // 12.6 MB
  __bf16* woutb = (__bf16*)carve((size_t)DMODEL * DMODEL * 2);     //  8.4 MB
  __bf16* qb    = (__bf16*)carve((size_t)NB * NH * S_LEN * HEADD * 2);
  __bf16* kb    = (__bf16*)carve((size_t)NB * NKV * S_LEN * HEADD * 2);
  __bf16* vTb   = (__bf16*)carve((size_t)NB * NKV * HEADD * S_LEN * 2);
  __bf16* ctxb  = (__bf16*)carve((size_t)M * DMODEL * 2);

  cvt_f32_bf16<<<2048, 256, 0, stream>>>(x, xb, M * DMODEL);
  cvt_f32_bf16<<<2048, 256, 0, stream>>>(Wqkv, wqkvb, QKVDIM * DMODEL);
  cvt_f32_bf16<<<2048, 256, 0, stream>>>(Wout, woutb, DMODEL * DMODEL);

  // QKV projection + bias + split/scatter (q, k, v-transposed) in bf16
  gemm_bf16<1><<<dim3(QKVDIM / 128, M / 128), 256, 0, stream>>>(
      xb, wqkvb, bqkv, nullptr, qb, kb, vTb, M, QKVDIM, DMODEL);

  // sliding-window GQA flash attention: 8192 wave-tasks / 4 waves per block
  attn_swa<<<2048, 128, 0, stream>>>(qb, kb, vTb, ctxb);

  // output projection + bias -> fp32 d_out
  gemm_bf16<0><<<dim3(DMODEL / 128, M / 128), 256, 0, stream>>>(
      ctxb, woutb, bout, out, nullptr, nullptr, nullptr, M, DMODEL, DMODEL);
}
